// AttnBlock_47107201302753
// MI455X (gfx1250) — matrix operand order repaired and checked by reading, not compiled, not run on hardware
//
#include <hip/hip_runtime.h>
#include <hip/hip_bf16.h>

// ---------------- CDNA5 WMMA plumbing (gfx1250, wave32) ----------------
typedef __attribute__((ext_vector_type(16))) __bf16 v16bf;
typedef __attribute__((ext_vector_type(8)))  float  v8f;
typedef __attribute__((ext_vector_type(4)))  int    i32x4;

union BF16Frag {
    v16bf v;
    unsigned int u[8];
    unsigned short h[16];
};

__device__ __forceinline__ unsigned short f2bf(float f) {
    unsigned u = __builtin_bit_cast(unsigned, f);
    unsigned r = u + 0x7fffu + ((u >> 16) & 1u);   // round-to-nearest-even
    return (unsigned short)(r >> 16);
}

__device__ __forceinline__ v8f wmma_bf16(const BF16Frag& a, const BF16Frag& b, v8f c) {
    // D = A(16x32 bf16) * B(32x16 bf16) + C(16x16 f32)
    return __builtin_amdgcn_wmma_f32_16x16x32_bf16(
        /*neg_a=*/false, a.v, /*neg_b=*/false, b.v,
        /*c_mod=*/(short)0, c, /*reuse_a=*/false, /*reuse_b=*/false);
}

// A-matrix (16x32 bf16): lane m=lane&15; VGPR i holds K-pair = half*16 + 2i
// lanes 0-15:  words {0..7} of the 16-word K window
// lanes 16-31: words {8..15}
__device__ __forceinline__ int a_word(int i, int half) {
    return half * 8 + i;
}
// B-matrix (32x16 bf16): lane n=lane&15; VGPR j holds K-pair = half*16 + 2j
__device__ __forceinline__ int b_word(int j, int half) {
    return half * 8 + j;
}

// ---------------- CDNA5 async global->LDS (ASYNCcnt) ----------------
#if defined(__has_builtin)
#if __has_builtin(__builtin_amdgcn_global_load_async_to_lds_b128) && \
    __has_builtin(__builtin_amdgcn_s_wait_asynccnt)
#define USE_ASYNC_LDS 1
#endif
#endif
#ifndef USE_ASYNC_LDS
#define USE_ASYNC_LDS 0
#endif

__device__ __forceinline__ void cp16_async(void* lds, const void* g) {
#if USE_ASYNC_LDS
    __builtin_amdgcn_global_load_async_to_lds_b128(
        (__attribute__((address_space(1))) i32x4*)g,
        (__attribute__((address_space(3))) i32x4*)lds, 0, 0);
#else
    *reinterpret_cast<int4*>(lds) = *reinterpret_cast<const int4*>(g);
#endif
}
__device__ __forceinline__ void wait_prev_async() {   // allow 8 in flight (next buffer)
#if USE_ASYNC_LDS
    __builtin_amdgcn_s_wait_asynccnt(8);
#endif
}
__device__ __forceinline__ void wait_all_async() {
#if USE_ASYNC_LDS
    __builtin_amdgcn_s_wait_asynccnt(0);
#endif
}

#define BATCH 8
#define CH    512
#define NPOS  4096
#define NGRP  32

// ---------------- weight f32 -> bf16 ----------------
__global__ void k_convert_w(const float* __restrict__ w, unsigned short* __restrict__ wbf, int n) {
    int i = blockIdx.x * blockDim.x + threadIdx.x;
    if (i < n) wbf[i] = f2bf(w[i]);
}

// ---------------- GroupNorm: x[B,C,N] f32 -> Hn[B,N,C] bf16 ----------------
__global__ void __launch_bounds__(256) k_groupnorm(
    const float* __restrict__ x, const float* __restrict__ gw,
    const float* __restrict__ gb, unsigned short* __restrict__ hn)
{
    int b = blockIdx.x / NGRP, g = blockIdx.x % NGRP;
    const float* xg = x + ((size_t)b * CH + g * 16) * NPOS;  // 16 contiguous channels

    float s1 = 0.f, s2 = 0.f;
    for (int idx = threadIdx.x; idx < 16 * NPOS; idx += 256) {
        float v = xg[idx]; s1 += v; s2 += v * v;
    }
    for (int m = 16; m >= 1; m >>= 1) { s1 += __shfl_xor(s1, m, 32); s2 += __shfl_xor(s2, m, 32); }
    __shared__ float ls1[8], ls2[8];
    int wid = threadIdx.x >> 5, lid = threadIdx.x & 31;
    if (lid == 0) { ls1[wid] = s1; ls2[wid] = s2; }
    __syncthreads();
    if (threadIdx.x == 0) {
        float a = 0.f, c = 0.f;
        for (int i = 0; i < 8; ++i) { a += ls1[i]; c += ls2[i]; }
        ls1[0] = a; ls2[0] = c;
    }
    __syncthreads();
    float mean = ls1[0] * (1.f / 65536.f);
    float var  = ls2[0] * (1.f / 65536.f) - mean * mean;
    float rstd = rsqrtf(var + 1e-6f);

    int cl = threadIdx.x & 15;            // channel in group
    int nr = threadIdx.x >> 4;            // 16 n-lanes
    int c  = g * 16 + cl;
    float gamma = gw[c] * rstd;
    float beta  = gb[c] - mean * gamma;   // y = x*gamma + beta
    const float* xc = x + ((size_t)b * CH + c) * NPOS;
    unsigned short* hrow = hn + (size_t)b * NPOS * CH + c;
    for (int n = nr; n < NPOS; n += 16)
        hrow[(size_t)n * CH] = f2bf(xc[n] * gamma + beta);
}

// ---------------- GEMM: Out[b,n,o] = (Hn[b,n,:] . W[o,:] + bias[o]) * scale, bf16 out ----------------
__global__ void __launch_bounds__(256) k_gemm_nc(
    const unsigned short* __restrict__ A,    // [B,N,C] bf16
    const unsigned short* __restrict__ Wbf,  // [C,C]   bf16 (row = out channel)
    const float* __restrict__ bias, float scale,
    unsigned short* __restrict__ Out)        // [B,N,C] bf16
{
    int b = blockIdx.z, nb = blockIdx.y * 64, ob = blockIdx.x * 64;
    int tid = threadIdx.x, w = tid >> 5, lane = tid & 31;
    int half = lane >> 4, l15 = lane & 15;
    int nsub = w & 3, opair = w >> 2;

    const unsigned int* Arow =
        (const unsigned int*)(A + ((size_t)b * NPOS + nb + nsub * 16 + l15) * CH);
    v8f acc[2] = {};

    for (int k = 0; k < 16; ++k) {              // K = 512 in steps of 32
        BF16Frag af;
        #pragma unroll
        for (int i = 0; i < 8; ++i) af.u[i] = Arow[k * 16 + a_word(i, half)];
        #pragma unroll
        for (int t = 0; t < 2; ++t) {
            int o = ob + (opair * 2 + t) * 16 + l15;
            const unsigned int* Wrow = (const unsigned int*)(Wbf + (size_t)o * CH);
            BF16Frag bf;
            #pragma unroll
            for (int j = 0; j < 8; ++j) bf.u[j] = Wrow[k * 16 + b_word(j, half)];
            acc[t] = wmma_bf16(af, bf, acc[t]);
        }
    }
    #pragma unroll
    for (int t = 0; t < 2; ++t) {
        int o = ob + (opair * 2 + t) * 16 + l15;
        float bv = bias[o];
        unsigned short* orow = Out + ((size_t)b * NPOS + nb + nsub * 16) * CH + o;
        #pragma unroll
        for (int r = 0; r < 8; ++r) {
            int m = r + half * 8;
            orow[(size_t)m * CH] = f2bf((acc[t][r] + bv) * scale);
        }
    }
}

// ---------------- Flash attention: Att[b,q,c] = softmax(Q Kt) V ----------------
__global__ void __launch_bounds__(256) k_attn(
    const unsigned short* __restrict__ Qn,  // [B,N,C] bf16 (pre-scaled by C^-0.5)
    const unsigned short* __restrict__ Kn,
    const unsigned short* __restrict__ Vn,
    unsigned short* __restrict__ Att)       // [B,N,C] bf16
{
    __shared__ float Sl[2][16][16];                 // shared score tiles (f32)
    __shared__ unsigned short Pl[8][16][32];        // per-wave P transpose staging
    __shared__ unsigned short Vl[2][32][512];       // double-buffered V key-block (2x32KB)

    int b = blockIdx.y, qb = blockIdx.x * 16;
    int tid = threadIdx.x, w = tid >> 5, lane = tid & 31;
    int half = lane >> 4, l15 = lane & 15;
    const size_t base = (size_t)b * NPOS * CH;

    // Q A-fragments for this wave's 64-channel slab (fixed for whole kernel)
    const unsigned int* Qrow =
        (const unsigned int*)(Qn + base + ((size_t)(qb + l15)) * CH) + w * 32;
    BF16Frag qa[2];
    #pragma unroll
    for (int cw = 0; cw < 2; ++cw)
        #pragma unroll
        for (int i = 0; i < 8; ++i) qa[cw].u[i] = Qrow[cw * 16 + a_word(i, half)];

    // prologue: async-stage first V key-block into buffer 0
    {
        const unsigned short* gsrc = Vn + base;
        #pragma unroll
        for (int i = 0; i < 8; ++i) {
            int c = i * 256 + tid;                  // 2048 x 16B chunks
            int krow = c >> 6, col = (c & 63) * 8;
            cp16_async(&Vl[0][krow][col], gsrc + (size_t)krow * CH + col);
        }
    }

    v8f acc[4] = {};
    float mrun[8], lrun[8], alpha[8];
    #pragma unroll
    for (int r = 0; r < 8; ++r) { mrun[r] = -1e30f; lrun[r] = 0.f; }

    for (int kb = 0; kb < NPOS; kb += 32) {
        int cur = (kb >> 5) & 1;

        // async-prefetch next V key-block into the other buffer, then wait for
        // the *current* buffer (in-order async completion: <=8 outstanding left)
        if (kb + 32 < NPOS) {
            const unsigned short* gsrc = Vn + base + (size_t)(kb + 32) * CH;
            #pragma unroll
            for (int i = 0; i < 8; ++i) {
                int c = i * 256 + tid;
                int krow = c >> 6, col = (c & 63) * 8;
                cp16_async(&Vl[cur ^ 1][krow][col], gsrc + (size_t)krow * CH + col);
            }
            wait_prev_async();
        } else {
            wait_all_async();
        }

        // zero shared score tiles (512 floats, 256 threads)
        ((float*)Sl)[tid] = 0.f;
        ((float*)Sl)[tid + 256] = 0.f;
        __syncthreads();

        // partial scores over this wave's 64-channel slab, two 16-key tiles
        #pragma unroll
        for (int t = 0; t < 2; ++t) {
            v8f s = {};
            int key = kb + t * 16 + l15;
            const unsigned int* Krow =
                (const unsigned int*)(Kn + base + (size_t)key * CH) + w * 32;
            #pragma unroll
            for (int cw = 0; cw < 2; ++cw) {
                BF16Frag kf;
                #pragma unroll
                for (int j = 0; j < 8; ++j) kf.u[j] = Krow[cw * 16 + b_word(j, half)];
                s = wmma_bf16(qa[cw], kf, s);
            }
            #pragma unroll
            for (int r = 0; r < 8; ++r)
                atomicAdd(&Sl[t][r + half * 8][l15], s[r]);   // ds_add_f32
        }
        __syncthreads();

        // online softmax over the 32-key block (lane = column; rows via VGPR idx)
        #pragma unroll
        for (int r = 0; r < 8; ++r) {
            int m = r + half * 8;
            float s0 = Sl[0][m][l15];
            float s1 = Sl[1][m][l15];
            float v = fmaxf(s0, s1);
            for (int msk = 8; msk >= 1; msk >>= 1) v = fmaxf(v, __shfl_xor(v, msk, 16));
            float mnew = fmaxf(mrun[r], v);
            float a = __expf(mrun[r] - mnew);
            float p0 = __expf(s0 - mnew);
            float p1 = __expf(s1 - mnew);
            float ps = p0 + p1;
            for (int msk = 8; msk >= 1; msk >>= 1) ps += __shfl_xor(ps, msk, 16);
            lrun[r] = lrun[r] * a + ps;
            mrun[r] = mnew;
            alpha[r] = a;
            Pl[w][m][l15]      = f2bf(p0);   // transpose staging (wave-private)
            Pl[w][m][16 + l15] = f2bf(p1);
        }

        // P A-fragment (16 queries x 32 keys) from LDS
        BF16Frag pa;
        const unsigned int* Prow = (const unsigned int*)&Pl[w][l15][0];
        #pragma unroll
        for (int i = 0; i < 8; ++i) pa.u[i] = Prow[a_word(i, half)];

        // PV: 4 channel tiles of 16 for this wave's slab (V from LDS), rescale + accumulate
        #pragma unroll
        for (int ct = 0; ct < 4; ++ct) {
            int ch = w * 64 + ct * 16 + l15;
            BF16Frag vf;
            #pragma unroll
            for (int j = 0; j < 8; ++j) {
                int kl = half * 16 + 2 * j;
                unsigned lo = Vl[cur][kl][ch];
                unsigned hi = Vl[cur][kl + 1][ch];
                vf.u[j] = lo | (hi << 16);
            }
            v8f a = acc[ct];
            #pragma unroll
            for (int r = 0; r < 8; ++r) a[r] *= alpha[r];
            acc[ct] = wmma_bf16(pa, vf, a);
        }
        __syncthreads();   // protect S tiles / V buffer before next iteration
    }

    // epilogue: normalize and store [B,N,C] bf16
    #pragma unroll
    for (int ct = 0; ct < 4; ++ct) {
        int ch = w * 64 + ct * 16 + l15;
        unsigned short* orow = Att + base + (size_t)qb * CH + ch;
        #pragma unroll
        for (int r = 0; r < 8; ++r) {
            int m = r + half * 8;
            orow[(size_t)m * CH] = f2bf(acc[ct][r] / lrun[r]);
        }
    }
}

// ---------------- proj GEMM + residual: out[b,o,n] f32 ----------------
__global__ void __launch_bounds__(256) k_proj(
    const unsigned short* __restrict__ A,    // Att [B,N,C] bf16
    const unsigned short* __restrict__ Wbf,  // proj_w bf16
    const float* __restrict__ bias,
    const float* __restrict__ x,             // residual [B,C,N] f32
    float* __restrict__ out)                 // [B,C,N] f32
{
    int b = blockIdx.z, nb = blockIdx.y * 64, ob = blockIdx.x * 64;
    int tid = threadIdx.x, w = tid >> 5, lane = tid & 31;
    int half = lane >> 4, l15 = lane & 15;
    int nsub = w & 3, opair = w >> 2;

    const unsigned int* Arow =
        (const unsigned int*)(A + ((size_t)b * NPOS + nb + nsub * 16 + l15) * CH);
    v8f acc[2] = {};

    for (int k = 0; k < 16; ++k) {
        BF16Frag af;
        #pragma unroll
        for (int i = 0; i < 8; ++i) af.u[i] = Arow[k * 16 + a_word(i, half)];
        #pragma unroll
        for (int t = 0; t < 2; ++t) {
            int o = ob + (opair * 2 + t) * 16 + l15;
            const unsigned int* Wrow = (const unsigned int*)(Wbf + (size_t)o * CH);
            BF16Frag bf;
            #pragma unroll
            for (int j = 0; j < 8; ++j) bf.u[j] = Wrow[k * 16 + b_word(j, half)];
            acc[t] = wmma_bf16(af, bf, acc[t]);
        }
    }
    #pragma unroll
    for (int t = 0; t < 2; ++t) {
        int o = ob + (opair * 2 + t) * 16 + l15;
        float bv = bias[o];
        const float* xr = x + ((size_t)b * CH + o) * NPOS + nb + nsub * 16;
        float* orow     = out + ((size_t)b * CH + o) * NPOS + nb + nsub * 16;
        #pragma unroll
        for (int r = 0; r < 8; ++r) {
            int m = r + half * 8;
            orow[m] = xr[m] + acc[t][r] + bv;
        }
    }
}

// ---------------- host side ----------------
extern "C" void kernel_launch(void* const* d_in, const int* in_sizes, int n_in,
                              void* d_out, int out_size, void* d_ws, size_t ws_size,
                              hipStream_t stream) {
    const float* x      = (const float*)d_in[0];
    const float* gn_w   = (const float*)d_in[1];
    const float* gn_b   = (const float*)d_in[2];
    const float* q_w    = (const float*)d_in[3];
    const float* q_b    = (const float*)d_in[4];
    const float* k_w    = (const float*)d_in[5];
    const float* k_b    = (const float*)d_in[6];
    const float* v_w    = (const float*)d_in[7];
    const float* v_b    = (const float*)d_in[8];
    const float* proj_w = (const float*)d_in[9];
    const float* proj_b = (const float*)d_in[10];
    float* out = (float*)d_out;

    const size_t WBYTES = (size_t)CH * CH * sizeof(unsigned short);           // 512 KB
    const size_t TBYTES = (size_t)BATCH * NPOS * CH * sizeof(unsigned short); // 32 MB
    char* ws = (char*)d_ws;
    unsigned short* wq = (unsigned short*)(ws);
    unsigned short* wk = (unsigned short*)(ws + WBYTES);
    unsigned short* wv = (unsigned short*)(ws + 2 * WBYTES);
    unsigned short* wp = (unsigned short*)(ws + 3 * WBYTES);
    unsigned short* Hn = (unsigned short*)(ws + 4 * WBYTES);
    unsigned short* Qn = (unsigned short*)(ws + 4 * WBYTES + TBYTES);
    unsigned short* Kn = (unsigned short*)(ws + 4 * WBYTES + 2 * TBYTES);
    unsigned short* Vn = (unsigned short*)(ws + 4 * WBYTES + 3 * TBYTES);
    unsigned short* At = (unsigned short*)(ws + 4 * WBYTES + 4 * TBYTES);

    const int WN = CH * CH;
    k_convert_w<<<(WN + 255) / 256, 256, 0, stream>>>(q_w, wq, WN);
    k_convert_w<<<(WN + 255) / 256, 256, 0, stream>>>(k_w, wk, WN);
    k_convert_w<<<(WN + 255) / 256, 256, 0, stream>>>(v_w, wv, WN);
    k_convert_w<<<(WN + 255) / 256, 256, 0, stream>>>(proj_w, wp, WN);

    k_groupnorm<<<BATCH * NGRP, 256, 0, stream>>>(x, gn_w, gn_b, Hn);

    dim3 ggrid(CH / 64, NPOS / 64, BATCH);
    float qscale = 1.0f / sqrtf((float)CH);
    k_gemm_nc<<<ggrid, 256, 0, stream>>>(Hn, wq, q_b, qscale, Qn);  // scale folded into Q
    k_gemm_nc<<<ggrid, 256, 0, stream>>>(Hn, wk, k_b, 1.0f, Kn);
    k_gemm_nc<<<ggrid, 256, 0, stream>>>(Hn, wv, v_b, 1.0f, Vn);

    dim3 agrid(NPOS / 16, BATCH);
    k_attn<<<agrid, 256, 0, stream>>>(Qn, Kn, Vn, At);

    k_proj<<<ggrid, 256, 0, stream>>>(At, wp, proj_b, x, out);
}
